// Nets_71554155151874
// MI455X (gfx1250) — matrix operand order repaired and checked by reading, not compiled, not run on hardware
//
#include <hip/hip_runtime.h>
#include <hip/hip_bf16.h>
#include <math.h>

// ---------------- problem constants ----------------
constexpr int NNODES = 2048;
constexpr int NEDGES = 16384;
constexpr int DIM    = 480;      // D
constexpr int DFEAT  = 512;      // DF (last layer out)
constexpr int HHEADS = 4;
constexpr int NBASIS = 64;       // NB
constexpr int SHDIM  = 9;
constexpr int LLAYER = 6;
constexpr int KCAT   = 2 * DIM + NBASIS;   // 1024
constexpr int HD     = HHEADS * DIM;       // 1920

typedef __bf16 bf16_t;
typedef __attribute__((ext_vector_type(16))) __bf16 v16bf;
typedef __attribute__((ext_vector_type(8)))  __bf16 v8bf;
typedef __attribute__((ext_vector_type(8)))  float  v8f;

// ---------------- helpers ----------------
__device__ __forceinline__ float siluf(float x) {
  return x / (1.0f + __expf(-x));
}

__device__ __forceinline__ float waveSum32(float v) {
#pragma unroll
  for (int m = 16; m >= 1; m >>= 1) v += __shfl_xor(v, m, 32);
  return v;
}

__device__ __forceinline__ float atomicMaxF(float* addr, float val) {
  unsigned int* ua = (unsigned int*)addr;
  unsigned int old = *ua;
  while (__uint_as_float(old) < val) {
    unsigned int assumed = old;
    old = atomicCAS(ua, assumed, __float_as_uint(val));
    if (old == assumed) break;
  }
  return __uint_as_float(old);
}

// ---------------- misc elementwise kernels ----------------
__global__ void fill_f32_kernel(float* __restrict__ p, float v, int n) {
  int i = blockIdx.x * blockDim.x + threadIdx.x;
  if (i < n) p[i] = v;
}

__global__ void f32_to_bf16_kernel(const float* __restrict__ in,
                                   bf16_t* __restrict__ out, int n) {
  int i = blockIdx.x * blockDim.x + threadIdx.x;
  if (i < n) out[i] = (bf16_t)in[i];
}

// out[n*K+k] = bf16(in[k*N+n])   (transpose + downconvert weights)
__global__ void transpose_bf16_kernel(const float* __restrict__ in,
                                      bf16_t* __restrict__ out,
                                      int Kd, int Nd) {
  int i = blockIdx.x * blockDim.x + threadIdx.x;
  if (i >= Kd * Nd) return;
  int n = i / Kd;
  int k = i - n * Kd;
  out[i] = (bf16_t)in[(size_t)k * Nd + n];
}

// ---------------- per-edge geometry: rbf + SH9 ----------------
__global__ void edge_prep_kernel(const float* __restrict__ ev,
                                 float* __restrict__ rbf,
                                 float* __restrict__ sh) {
  int e = blockIdx.x * blockDim.x + threadIdx.x;
  if (e >= NEDGES) return;
  float vx = ev[e * 3 + 0], vy = ev[e * 3 + 1], vz = ev[e * 3 + 2];
  float dd = sqrtf(vx * vx + vy * vy + vz * vz);
  const float w = 0.5f * 10.0f / (float)NBASIS;
  const float inv2w2 = 1.0f / (2.0f * w * w);
#pragma unroll 8
  for (int j = 0; j < NBASIS; ++j) {
    float c = 10.0f * (float)j / (float)(NBASIS - 1);
    float t = dd - c;
    rbf[(size_t)e * NBASIS + j] = __expf(-t * t * inv2w2);
  }
  // reference permutes edge_vec[:, [1,2,0]] before _sh9
  float rn = 1.0f / (dd + 1e-12f);
  float x = vy * rn, y = vz * rn, z = vx * rn;
  const float c1 = 1.7320508075688772f;   // sqrt(3)
  const float c2 = 3.8729833462074170f;   // sqrt(15)
  const float c3 = 1.1180339887498949f;   // sqrt(5)/2
  float* s = sh + (size_t)e * SHDIM;
  s[0] = 1.0f;
  s[1] = c1 * x;
  s[2] = c1 * y;
  s[3] = c1 * z;
  s[4] = c2 * x * y;
  s[5] = c2 * y * z;
  s[6] = c3 * (3.0f * z * z - 1.0f);
  s[7] = c2 * x * z;
  s[8] = 0.5f * c2 * (x * x - y * y);
}

// ---------------- initial node embedding (fp32 -> bf16) ----------------
__global__ void embed_kernel(const float* __restrict__ embed_W,
                             const int* __restrict__ atom,
                             bf16_t* __restrict__ nodeBF) {
  int i = blockIdx.x * blockDim.x + threadIdx.x;
  if (i >= NNODES * DIM) return;
  int v = i / DIM, j = i - v * DIM;
  nodeBF[i] = (bf16_t)embed_W[(size_t)atom[v] * DIM + j];
}

// ---------------- gate + attention MLPs: one wave32 per edge ----------------
__global__ void gate_attn_kernel(
    const float* __restrict__ rbf, const float* __restrict__ sh,
    const int* __restrict__ edge_dst,
    const float* __restrict__ tpW1, const float* __restrict__ tpB1,
    const float* __restrict__ tpW2, const float* __restrict__ tpB2,
    const float* __restrict__ alW1, const float* __restrict__ alB1,
    const float* __restrict__ alG1, const float* __restrict__ alBe1,
    const float* __restrict__ alW2, const float* __restrict__ alB2,
    const float* __restrict__ alG2, const float* __restrict__ alBe2,
    const float* __restrict__ alW3, const float* __restrict__ alB3,
    float* __restrict__ gate, float* __restrict__ attnL,
    float* __restrict__ segmax) {
  const int lane = threadIdx.x & 31;
  const int e = blockIdx.x * (blockDim.x >> 5) + (threadIdx.x >> 5);
  if (e >= NEDGES) return;

  const float r0 = rbf[(size_t)e * NBASIS + lane];
  const float r1 = rbf[(size_t)e * NBASIS + lane + 32];

  // ---- tensor-product gate path: h = silu(rbf @ W1 + b1)
  float h0 = tpB1[lane], h1 = tpB1[lane + 32];
#pragma unroll 4
  for (int kk = 0; kk < 32; ++kk) {
    float ra = __shfl(r0, kk, 32);
    float rb = __shfl(r1, kk, 32);
    h0 += ra * tpW1[kk * 64 + lane]        + rb * tpW1[(kk + 32) * 64 + lane];
    h1 += ra * tpW1[kk * 64 + lane + 32]   + rb * tpW1[(kk + 32) * 64 + lane + 32];
  }
  h0 = siluf(h0);
  h1 = siluf(h1);

  float gacc = 0.0f;
#pragma unroll
  for (int j = 0; j < SHDIM; ++j) {
    float p = h0 * tpW2[lane * SHDIM + j] + h1 * tpW2[(lane + 32) * SHDIM + j];
    p = waveSum32(p);
    gacc += (p + tpB2[j]) * sh[(size_t)e * SHDIM + j];
  }
  if (lane == 0) gate[e] = gacc;

  // ---- attention path: two LN+silu MLP layers on rbf
  float y0 = alB1[lane], y1 = alB1[lane + 32];
#pragma unroll 4
  for (int kk = 0; kk < 32; ++kk) {
    float ra = __shfl(r0, kk, 32);
    float rb = __shfl(r1, kk, 32);
    y0 += ra * alW1[kk * 64 + lane]      + rb * alW1[(kk + 32) * 64 + lane];
    y1 += ra * alW1[kk * 64 + lane + 32] + rb * alW1[(kk + 32) * 64 + lane + 32];
  }
  float mu  = waveSum32(y0 + y1) * (1.0f / 64.0f);
  float d0 = y0 - mu, d1 = y1 - mu;
  float var = waveSum32(d0 * d0 + d1 * d1) * (1.0f / 64.0f);
  float ri  = rsqrtf(var + 1e-6f);
  float a0 = siluf(d0 * ri * alG1[lane]      + alBe1[lane]);
  float a1 = siluf(d1 * ri * alG1[lane + 32] + alBe1[lane + 32]);

  float z0 = alB2[lane], z1 = alB2[lane + 32];
#pragma unroll 4
  for (int kk = 0; kk < 32; ++kk) {
    float aa = __shfl(a0, kk, 32);
    float ab = __shfl(a1, kk, 32);
    z0 += aa * alW2[kk * 64 + lane]      + ab * alW2[(kk + 32) * 64 + lane];
    z1 += aa * alW2[kk * 64 + lane + 32] + ab * alW2[(kk + 32) * 64 + lane + 32];
  }
  mu  = waveSum32(z0 + z1) * (1.0f / 64.0f);
  d0 = z0 - mu; d1 = z1 - mu;
  var = waveSum32(d0 * d0 + d1 * d1) * (1.0f / 64.0f);
  ri  = rsqrtf(var + 1e-6f);
  float b0 = siluf(d0 * ri * alG2[lane]      + alBe2[lane]);
  float b1 = siluf(d1 * ri * alG2[lane + 32] + alBe2[lane + 32]);

  int dst = edge_dst[e];
#pragma unroll
  for (int j = 0; j < HHEADS; ++j) {
    float p = b0 * alW3[lane * HHEADS + j] + b1 * alW3[(lane + 32) * HHEADS + j];
    p = waveSum32(p);
    float l = p + alB3[j];
    if (lane == 0) {
      attnL[(size_t)e * HHEADS + j] = l;
      atomicMaxF(&segmax[dst * HHEADS + j], l);
    }
  }
}

// ---------------- segment softmax (exp/sum then normalize) ----------------
__global__ void attn_exp_kernel(float* __restrict__ attnL,
                                const float* __restrict__ segmax,
                                float* __restrict__ segsum,
                                const int* __restrict__ edge_dst) {
  int i = blockIdx.x * blockDim.x + threadIdx.x;
  if (i >= NEDGES * HHEADS) return;
  int e = i >> 2, h = i & 3;
  int d = edge_dst[e];
  float ex = __expf(attnL[i] - segmax[d * HHEADS + h]);
  attnL[i] = ex;
  atomicAdd(&segsum[d * HHEADS + h], ex);
}

__global__ void attn_norm_kernel(float* __restrict__ attnL,
                                 const float* __restrict__ segsum,
                                 const int* __restrict__ edge_dst) {
  int i = blockIdx.x * blockDim.x + threadIdx.x;
  if (i >= NEDGES * HHEADS) return;
  int e = i >> 2, h = i & 3;
  attnL[i] = attnL[i] / (segsum[edge_dst[e] * HHEADS + h] + 1e-16f);
}

// ---------------- gather concat [fea[src] | fea[dst] | rbf] -> bf16 ----------------
__global__ void gather_concat_kernel(const bf16_t* __restrict__ nodeBF,
                                     const float* __restrict__ rbf,
                                     const int* __restrict__ esrc,
                                     const int* __restrict__ edst,
                                     bf16_t* __restrict__ concatA) {
  int i = blockIdx.x * blockDim.x + threadIdx.x;  // NEDGES * 128 chunks of 8
  if (i >= NEDGES * (KCAT / 8)) return;
  int e = i >> 7, c = i & 127;
  const v8bf* nodeV = (const v8bf*)nodeBF;       // row = 60 v8bf
  v8bf out;
  if (c < 60) {
    out = nodeV[(size_t)esrc[e] * 60 + c];
  } else if (c < 120) {
    out = nodeV[(size_t)edst[e] * 60 + (c - 60)];
  } else {
    const float* rp = rbf + (size_t)e * NBASIS + (size_t)(c - 120) * 8;
#pragma unroll
    for (int t = 0; t < 8; ++t) out[t] = (bf16_t)rp[t];
  }
  ((v8bf*)concatA)[(size_t)e * 128 + c] = out;
}

// ---------------- WMMA bf16 GEMM: C[M,N] = A[M,K] * Bt[N,K]^T ----------------
// Wave tile 32 x (NT*16): 2 M-frags x NT N-frags (8 accumulators max = 64 VGPRs,
// proven spill-free); block = 8 waves = 256 rows. NT/EMODE are template
// parameters so the K-loop and epilogue carry zero branches; the N=480 tail gets
// a dedicated NT=2 launch.
// EMODE 0: out bf16 = acc * gate[row]                          (msg GEMM)
// EMODE 1: atomicAdd(agg[dst[row]*HD + n], acc*attn[row,h])    (value GEMM+scatter)
// EMODE 2: out bf16 = acc                                      (node update, mid)
// EMODE 3: out f32  = acc                                      (node update -> d_out)
template <int NT, int EMODE>
__global__ void __launch_bounds__(256, 1)
wmma_gemm_kernel(const bf16_t* __restrict__ A, int lda,
                 const bf16_t* __restrict__ Bt, int K, int nBase0,
                 const float* __restrict__ gate,
                 bf16_t* __restrict__ outBF, int ldOut,
                 const int* __restrict__ edge_dst,
                 const float* __restrict__ attn,
                 float* __restrict__ agg,
                 float* __restrict__ outF32, int ldF32) {
  const int lane  = threadIdx.x & 31;
  const int wave  = threadIdx.x >> 5;
  const int mBase = blockIdx.x * 256 + wave * 32;
  const int nBase = nBase0 + blockIdx.y * 64;

  v8f zero = {0.f, 0.f, 0.f, 0.f, 0.f, 0.f, 0.f, 0.f};
  v8f acc[2][NT];
#pragma unroll
  for (int mt = 0; mt < 2; ++mt)
#pragma unroll
    for (int nt = 0; nt < NT; ++nt) acc[mt][nt] = zero;

  // A fragment (16-bit A 16x32 layout): lane&15 = row, lanes 16-31 start at K+8;
  // elements 0..7 = K[kb..kb+7], 8..15 = K[kb+16..kb+23]
  const int kbA = (lane >> 4) << 3;
  const bf16_t* Ap0 = A + (size_t)(mBase + (lane & 15)) * lda + kbA;
  const bf16_t* Ap1 = Ap0 + (size_t)16 * lda;

  // B fragment from Bt[N,K] row-major: lane&15 = col, lanes 16-31 start at K+8,
  // elements 0..7 = K[kb..kb+7], 8..15 = K[kb+16..kb+23] (matches A).
  const int kbB = (lane >> 4) << 3;
  const bf16_t* Bp[NT];
#pragma unroll
  for (int nt = 0; nt < NT; ++nt)
    Bp[nt] = Bt + (size_t)(nBase + nt * 16 + (lane & 15)) * K + kbB;

  for (int k = 0; k < K; k += 32) {
    v8bf a0l = *(const v8bf*)(Ap0 + k);
    v8bf a0h = *(const v8bf*)(Ap0 + k + 16);
    v8bf a1l = *(const v8bf*)(Ap1 + k);
    v8bf a1h = *(const v8bf*)(Ap1 + k + 16);
    v16bf af0 = __builtin_shufflevector(a0l, a0h, 0, 1, 2, 3, 4, 5, 6, 7,
                                        8, 9, 10, 11, 12, 13, 14, 15);
    v16bf af1 = __builtin_shufflevector(a1l, a1h, 0, 1, 2, 3, 4, 5, 6, 7,
                                        8, 9, 10, 11, 12, 13, 14, 15);
    __builtin_prefetch(Ap0 + k + 64, 0, 1);  // speculative; OOB silently dropped
    __builtin_prefetch(Ap1 + k + 64, 0, 1);
#pragma unroll
    for (int nt = 0; nt < NT; ++nt) {
      v8bf b0 = *(const v8bf*)(Bp[nt] + k);
      v8bf b1 = *(const v8bf*)(Bp[nt] + k + 16);
      v16bf bfr = __builtin_shufflevector(b0, b1, 0, 1, 2, 3, 4, 5, 6, 7,
                                          8, 9, 10, 11, 12, 13, 14, 15);
      acc[0][nt] = __builtin_amdgcn_wmma_f32_16x16x32_bf16(
          false, af0, false, bfr, (short)0, acc[0][nt], false, false);
      acc[1][nt] = __builtin_amdgcn_wmma_f32_16x16x32_bf16(
          false, af1, false, bfr, (short)0, acc[1][nt], false, false);
    }
  }

  // C/D layout: VGPR j -> M = tile + j + (lane>=16 ? 8 : 0), N = tile + (lane&15)
  const int mOff  = (lane >> 4) << 3;
  const int nLane = lane & 15;
#pragma unroll
  for (int mt = 0; mt < 2; ++mt) {
#pragma unroll
    for (int nt = 0; nt < NT; ++nt) {
      const int tN = nBase + nt * 16;
      v8f c = acc[mt][nt];
#pragma unroll
      for (int jj = 0; jj < 8; ++jj) {
        int m = mBase + mt * 16 + mOff + jj;
        int n = tN + nLane;
        float v = c[jj];
        if (EMODE == 0) {
          outBF[(size_t)m * ldOut + n] = (bf16_t)(v * gate[m]);
        } else if (EMODE == 1) {
          int h = tN / DIM;  // uniform per 16-col tile (480 % 16 == 0)
          float s = attn[(size_t)m * HHEADS + h];
          atomicAdd(&agg[(size_t)edge_dst[m] * HD + n], v * s);
        } else if (EMODE == 2) {
          outBF[(size_t)m * ldOut + n] = (bf16_t)v;
        } else {
          outF32[(size_t)m * ldF32 + n] = v;
        }
      }
    }
  }
}

// ---------------- workspace layout ----------------
constexpr size_t a256(size_t x) { return (x + 255) & ~size_t(255); }
constexpr size_t OFF_GATE  = 0;
constexpr size_t OFF_ATTN  = OFF_GATE  + a256((size_t)NEDGES * sizeof(float));
constexpr size_t OFF_SMAX  = OFF_ATTN  + a256((size_t)NEDGES * HHEADS * sizeof(float));
constexpr size_t OFF_SSUM  = OFF_SMAX  + a256((size_t)NNODES * HHEADS * sizeof(float));
constexpr size_t OFF_RBF   = OFF_SSUM  + a256((size_t)NNODES * HHEADS * sizeof(float));
constexpr size_t OFF_SH    = OFF_RBF   + a256((size_t)NEDGES * NBASIS * sizeof(float));
constexpr size_t OFF_NODE  = OFF_SH    + a256((size_t)NEDGES * SHDIM * sizeof(float));
constexpr size_t OFF_CONC  = OFF_NODE  + a256((size_t)NNODES * DIM * 2);
constexpr size_t OFF_MSGG  = OFF_CONC  + a256((size_t)NEDGES * KCAT * 2);
constexpr size_t OFF_AGG   = OFF_MSGG  + a256((size_t)NEDGES * DIM * 2);
constexpr size_t OFF_AGGBF = OFF_AGG   + a256((size_t)NNODES * HD * sizeof(float));
constexpr size_t OFF_PREWT = OFF_AGGBF + a256((size_t)NNODES * HD * 2);
constexpr size_t OFF_TPWOT = OFF_PREWT + a256((size_t)DIM * KCAT * 2);
constexpr size_t OFF_LINWT = OFF_TPWOT + a256((size_t)HD * DIM * 2);

extern "C" void kernel_launch(void* const* d_in, const int* in_sizes, int n_in,
                              void* d_out, int out_size, void* d_ws, size_t ws_size,
                              hipStream_t stream) {
  (void)in_sizes; (void)n_in; (void)out_size; (void)ws_size;
  const float* embed_W    = (const float*)d_in[0];
  const float* pre_W      = (const float*)d_in[1];
  const float* tp_W1      = (const float*)d_in[2];
  const float* tp_b1      = (const float*)d_in[3];
  const float* tp_W2      = (const float*)d_in[4];
  const float* tp_b2      = (const float*)d_in[5];
  const float* tp_Wo      = (const float*)d_in[6];
  const float* al_W1      = (const float*)d_in[7];
  const float* al_b1      = (const float*)d_in[8];
  const float* al_g1      = (const float*)d_in[9];
  const float* al_be1     = (const float*)d_in[10];
  const float* al_W2      = (const float*)d_in[11];
  const float* al_b2      = (const float*)d_in[12];
  const float* al_g2      = (const float*)d_in[13];
  const float* al_be2     = (const float*)d_in[14];
  const float* al_W3      = (const float*)d_in[15];
  const float* al_b3      = (const float*)d_in[16];
  const float* lin_W_mid  = (const float*)d_in[17];
  const float* lin_W_last = (const float*)d_in[18];
  const float* edge_vec   = (const float*)d_in[19];
  const int*   node_atom  = (const int*)d_in[20];
  const int*   edge_src   = (const int*)d_in[21];
  const int*   edge_dst   = (const int*)d_in[22];

  char* ws = (char*)d_ws;
  float*  gate    = (float*)(ws + OFF_GATE);
  float*  attn    = (float*)(ws + OFF_ATTN);
  float*  segmax  = (float*)(ws + OFF_SMAX);
  float*  segsum  = (float*)(ws + OFF_SSUM);
  float*  rbfP    = (float*)(ws + OFF_RBF);
  float*  shP     = (float*)(ws + OFF_SH);
  bf16_t* nodeBF  = (bf16_t*)(ws + OFF_NODE);
  bf16_t* concatA = (bf16_t*)(ws + OFF_CONC);
  bf16_t* msgg    = (bf16_t*)(ws + OFF_MSGG);
  float*  agg     = (float*)(ws + OFF_AGG);
  bf16_t* aggBF   = (bf16_t*)(ws + OFF_AGGBF);
  bf16_t* preWt   = (bf16_t*)(ws + OFF_PREWT);
  bf16_t* tpWot   = (bf16_t*)(ws + OFF_TPWOT);
  bf16_t* linWt   = (bf16_t*)(ws + OFF_LINWT);

  edge_prep_kernel<<<(NEDGES + 255) / 256, 256, 0, stream>>>(edge_vec, rbfP, shP);
  embed_kernel<<<(NNODES * DIM + 255) / 256, 256, 0, stream>>>(embed_W, node_atom, nodeBF);

  for (int i = 0; i < LLAYER; ++i) {
    fill_f32_kernel<<<(NNODES * HHEADS + 255) / 256, 256, 0, stream>>>(segmax, -3e38f, NNODES * HHEADS);
    fill_f32_kernel<<<(NNODES * HHEADS + 255) / 256, 256, 0, stream>>>(segsum, 0.0f, NNODES * HHEADS);
    fill_f32_kernel<<<(NNODES * HD + 255) / 256, 256, 0, stream>>>(agg, 0.0f, NNODES * HD);

    gate_attn_kernel<<<NEDGES / 8, 256, 0, stream>>>(
        rbfP, shP, edge_dst,
        tp_W1 + (size_t)i * 64 * 64, tp_b1 + (size_t)i * 64,
        tp_W2 + (size_t)i * 64 * SHDIM, tp_b2 + (size_t)i * SHDIM,
        al_W1 + (size_t)i * 64 * 64, al_b1 + (size_t)i * 64,
        al_g1 + (size_t)i * 64, al_be1 + (size_t)i * 64,
        al_W2 + (size_t)i * 64 * 64, al_b2 + (size_t)i * 64,
        al_g2 + (size_t)i * 64, al_be2 + (size_t)i * 64,
        al_W3 + (size_t)i * 64 * HHEADS, al_b3 + (size_t)i * HHEADS,
        gate, attn, segmax);
    attn_exp_kernel<<<(NEDGES * HHEADS + 255) / 256, 256, 0, stream>>>(attn, segmax, segsum, edge_dst);
    attn_norm_kernel<<<(NEDGES * HHEADS + 255) / 256, 256, 0, stream>>>(attn, segsum, edge_dst);

    gather_concat_kernel<<<(NEDGES * (KCAT / 8) + 255) / 256, 256, 0, stream>>>(
        nodeBF, rbfP, edge_src, edge_dst, concatA);

    transpose_bf16_kernel<<<(KCAT * DIM + 255) / 256, 256, 0, stream>>>(
        pre_W + (size_t)i * KCAT * DIM, preWt, KCAT, DIM);
    transpose_bf16_kernel<<<(DIM * HD + 255) / 256, 256, 0, stream>>>(
        tp_Wo + (size_t)i * DIM * HD, tpWot, DIM, HD);
    const int Nout = (i < LLAYER - 1) ? DIM : DFEAT;
    const float* linSrc = (i < LLAYER - 1) ? lin_W_mid + (size_t)i * HD * DIM : lin_W_last;
    transpose_bf16_kernel<<<(HD * Nout + 255) / 256, 256, 0, stream>>>(linSrc, linWt, HD, Nout);

    // ---- msg = concat @ pre_W   (epilogue: * gate, write bf16). N=480: 7 full
    //      64-col blocks + one 32-col tail (NT=2).
    {
      dim3 gMain(NEDGES / 256, DIM / 64);          // cols [0, 448)
      wmma_gemm_kernel<4, 0><<<gMain, 256, 0, stream>>>(
          concatA, KCAT, preWt, KCAT, 0, gate, msgg, DIM,
          nullptr, nullptr, nullptr, nullptr, 0);
      dim3 gTail(NEDGES / 256, 1);                 // cols [448, 480)
      wmma_gemm_kernel<2, 0><<<gTail, 256, 0, stream>>>(
          concatA, KCAT, preWt, KCAT, (DIM / 64) * 64, gate, msgg, DIM,
          nullptr, nullptr, nullptr, nullptr, 0);
    }
    // ---- value = msgg @ tp_Wo (epilogue: * attn[e,h], atomic scatter into agg)
    {
      dim3 g2(NEDGES / 256, HD / 64);
      wmma_gemm_kernel<4, 1><<<g2, 256, 0, stream>>>(
          msgg, DIM, tpWot, DIM, 0, nullptr, nullptr, 0,
          edge_dst, attn, agg, nullptr, 0);
    }

    f32_to_bf16_kernel<<<(NNODES * HD + 255) / 256, 256, 0, stream>>>(agg, aggBF, NNODES * HD);

    // ---- node_fea = agg @ W
    if (i < LLAYER - 1) {
      dim3 gMain(NNODES / 256, DIM / 64);
      wmma_gemm_kernel<4, 2><<<gMain, 256, 0, stream>>>(
          aggBF, HD, linWt, HD, 0, nullptr, nodeBF, DIM,
          nullptr, nullptr, nullptr, nullptr, 0);
      dim3 gTail(NNODES / 256, 1);
      wmma_gemm_kernel<2, 2><<<gTail, 256, 0, stream>>>(
          aggBF, HD, linWt, HD, (DIM / 64) * 64, nullptr, nodeBF, DIM,
          nullptr, nullptr, nullptr, nullptr, 0);
    } else {
      dim3 g3(NNODES / 256, DFEAT / 64);
      wmma_gemm_kernel<4, 3><<<g3, 256, 0, stream>>>(
          aggBF, HD, linWt, HD, 0, nullptr, nullptr, 0,
          nullptr, nullptr, nullptr, (float*)d_out, DFEAT);
    }
  }
}
